// DecoderAttention_56401510531619
// MI455X (gfx1250) — compile-verified, matrix operand order checked by reading, not run on hardware
//
#include <hip/hip_runtime.h>

#define NH   16
#define DM   1024
#define DH   64
#define PSEQ 2048
#define NB   2

typedef __bf16 v16bf __attribute__((ext_vector_type(16)));
typedef float  v8f   __attribute__((ext_vector_type(8)));
typedef unsigned int v4u __attribute__((ext_vector_type(4)));
typedef int          v8i __attribute__((ext_vector_type(8)));
typedef int          v4i __attribute__((ext_vector_type(4)));

#if defined(__HIP_DEVICE_COMPILE__) && __has_builtin(__builtin_amdgcn_tensor_load_to_lds) && __has_builtin(__builtin_amdgcn_s_wait_tensorcnt)
#define HAVE_TDM 1
#else
#define HAVE_TDM 0
#endif

struct B32x8 { uint4 lo, hi; };

static __device__ __forceinline__ unsigned bfbits(float f) {
  unsigned u = __float_as_uint(f);
  return (u + 0x7FFFu + ((u >> 16) & 1u)) >> 16;   // RNE f32 -> bf16
}
static __device__ __forceinline__ unsigned pk2(float a, float b) {
  return bfbits(a) | (bfbits(b) << 16);
}

// 16x32 (A: lane=M) or 32x16 (B: lane=N) bf16 fragment from row-major memory,
// K contiguous. CDNA5 ISA 7.12.2: lanes 0-15 -> K 0..7 & 16..23,
// lanes 16-31 -> K 8..15 & 24..31, two bf16 per VGPR.
static __device__ __forceinline__ v16bf load_frag_b16(const unsigned short* base, int stride) {
  const int lane = threadIdx.x & 31;
  const int r    = lane & 15;
  const int koff = (lane & 16) ? 8 : 0;
  const unsigned short* p = base + (size_t)r * stride + koff;
  B32x8 s;
  s.lo = *(const uint4*)(p);
  s.hi = *(const uint4*)(p + 16);
  return __builtin_bit_cast(v16bf, s);
}

#define WMMA(a,b,c) __builtin_amdgcn_wmma_f32_16x16x32_bf16(false,(a),false,(b),(short)0,(c),false,false)

static __device__ __forceinline__ unsigned lds_off(const void* p) {
  return (unsigned)(unsigned long long)p;   // generic LDS addr: low 32 bits = LDS byte offset
}

#if HAVE_TDM
// TDM: DMA a 64-row x 64-col bf16 tile (row stride row_stride_elems) into LDS.
// pad_enable: every 32 DWORDs (one 128B row) insert 4 DWORDs -> LDS row stride
// 72 ushorts (144B), conflict-free for 16-lane b128 fragment reads.
static __device__ __forceinline__ void tdm_load_tile(unsigned lds_byte_off, const void* gaddr,
                                                     unsigned row_stride_elems) {
  unsigned long long ga = (unsigned long long)gaddr;
  v4u g0 = { 1u,                                            // count=1, user mode
             lds_byte_off,
             (unsigned)ga,
             (unsigned)((ga >> 32) & 0x1ffffffu) | (2u << 30) };   // addr[56:32] | type=2
  v8i g1;
  g1[0] = (int)((1u << 16) | (1u << 20) | (4u << 22) | (3u << 25)); // data_size=2B, pad: 32 DW + 4 DW
  g1[1] = (int)(64u << 16);          // tensor_dim0 = 64
  g1[2] = (int)(2048u << 16);        // tensor_dim1 = 2048
  g1[3] = (int)(64u << 16);          // tile_dim0 = 64
  g1[4] = 64;                        // tile_dim1 = 64, tile_dim2 = 0
  g1[5] = (int)row_stride_elems;     // tensor_dim0_stride
  g1[6] = 0;
  g1[7] = 0;
  v4i z4 = {0, 0, 0, 0};
#if __clang_major__ >= 23
  v8i z8 = {0, 0, 0, 0, 0, 0, 0, 0};
  __builtin_amdgcn_tensor_load_to_lds(g0, g1, z4, z4, z8, 0);
#else
  __builtin_amdgcn_tensor_load_to_lds(g0, g1, z4, z4, 0);
#endif
}
#else
// Fallback: cooperative staged copy (256 threads), same padded LDS layout.
static __device__ __forceinline__ void stage_tile(unsigned short* dst, const unsigned short* src,
                                                  int src_stride) {
  const int t = threadIdx.x;
  const int row = t >> 2, c = (t & 3) * 16;
  uint4 a = *(const uint4*)(src + (size_t)row * src_stride + c);
  uint4 b = *(const uint4*)(src + (size_t)row * src_stride + c + 8);
  *(uint4*)(dst + row * 72 + c) = a;
  *(uint4*)(dst + row * 72 + c + 8) = b;
}
#endif

// -------- f32 -> bf16 bulk convert (8 elems/thread) --------
__global__ __launch_bounds__(256)
void k_cvt_bf16(const float* __restrict__ src, unsigned short* __restrict__ dst) {
  const int i = blockIdx.x * 256 + threadIdx.x;
  const float4* s = (const float4*)src;
  float4 a = s[2 * i], b = s[2 * i + 1];
  ((uint4*)dst)[i] = make_uint4(pk2(a.x, a.y), pk2(a.z, a.w), pk2(b.x, b.y), pk2(b.z, b.w));
}

// -------- transpose + f32->bf16 convert:  src [z][R][C] -> dst [z][C][R] --------
__global__ __launch_bounds__(256)
void k_transpose_cvt(const float* __restrict__ src, unsigned short* __restrict__ dst,
                     int R, int C) {
  __shared__ float tile[32][33];
  const size_t boff = (size_t)blockIdx.z * R * C;
  src += boff; dst += boff;
  const int c0 = blockIdx.x * 32, r0 = blockIdx.y * 32;
  const int tx = threadIdx.x, ty = threadIdx.y;
#pragma unroll
  for (int i = 0; i < 32; i += 8) {
    int rr = r0 + ty + i, cc = c0 + tx;
    tile[ty + i][tx] = (rr < R && cc < C) ? src[(size_t)rr * C + cc] : 0.f;
  }
  __syncthreads();
#pragma unroll
  for (int i = 0; i < 32; i += 8) {
    int rr = c0 + ty + i, cc = r0 + tx;
    if (rr < C && cc < R) dst[(size_t)rr * R + cc] = (unsigned short)bfbits(tile[tx][ty + i]);
  }
}

// -------- projection GEMM: Y[m, h*64+dk] = sum_d X[m,d] * W[h,d,dk] + bias --------
// X bf16 [m][d]; WT bf16 [h][dk][d]. storeT=1 stores V transposed as [b,h,dk,p].
__global__ __launch_bounds__(256)
void k_proj(const unsigned short* __restrict__ X, const unsigned short* __restrict__ WT,
            const float* __restrict__ bias, unsigned short* __restrict__ out, int storeT) {
  const int h    = blockIdx.y;
  const int wave = threadIdx.x >> 5, lane = threadIdx.x & 31;
  const int r    = lane & 15,        half = lane >> 4;
  const int m0   = blockIdx.x * 128 + wave * 16;
  const unsigned short* wt = WT + (size_t)h * (DH * DM);
  v8f zv = {};
  v8f acc[4] = {zv, zv, zv, zv};
  for (int k0 = 0; k0 < DM; k0 += 32) {
    v16bf a = load_frag_b16(X + (size_t)m0 * DM + k0, DM);
#pragma unroll
    for (int n = 0; n < 4; ++n) {
      v16bf b = load_frag_b16(wt + (size_t)(n * 16) * DM + k0, DM);
      acc[n] = WMMA(a, b, acc[n]);
    }
  }
  const int bb = m0 / PSEQ, p0 = m0 - bb * PSEQ;
#pragma unroll
  for (int n = 0; n < 4; ++n) {
    const int dk = n * 16 + r;
    const float bv = bias[h * DH + dk];
#pragma unroll
    for (int rr = 0; rr < 8; ++rr) {
      const int row = rr + half * 8;
      const unsigned short bf = (unsigned short)bfbits(acc[n][rr] + bv);
      if (!storeT)
        out[(size_t)(m0 + row) * DM + h * DH + dk] = bf;
      else
        out[((size_t)((bb * NH + h) * DH + dk)) * PSEQ + (p0 + row)] = bf;
    }
  }
}

// -------- attention core --------
// qw,kw: [b,p,h,dk] bf16 ; vT: [b,h,dk,p] bf16 ; zout: [b,p,h,dk] bf16
// pat: [b,h,q,k] f32. K/V tiles TDM-staged to LDS, shared by all 8 waves.
__global__ __launch_bounds__(256)
void k_attn(const unsigned short* __restrict__ qw, const unsigned short* __restrict__ kw,
            const unsigned short* __restrict__ vT, unsigned short* __restrict__ zout,
            float* __restrict__ pat) {
  __shared__ unsigned short kstage[2][64 * 72];
  __shared__ unsigned short vstage[2][64 * 72];
  __shared__ unsigned short plds[8][16][88];

  const int bh = blockIdx.y, bb = bh >> 4, h = bh & 15;
  const int wave = threadIdx.x >> 5, lane = threadIdx.x & 31;
  const int r = lane & 15, half = lane >> 4;
  const int q0blk = blockIdx.x * 128;
  const int q0 = q0blk + wave * 16;

  const unsigned short* qb = qw + (size_t)bb * PSEQ * DM + h * DH;
  const unsigned short* kb = kw + (size_t)bb * PSEQ * DM + h * DH;
  const unsigned short* vb = vT + (size_t)bh * DH * PSEQ;
  float* patb = pat + ((size_t)bh * PSEQ + q0) * PSEQ;

  const v16bf qlo = load_frag_b16(qb + (size_t)q0 * DM, DM);        // dk 0..31
  const v16bf qhi = load_frag_b16(qb + (size_t)q0 * DM + 32, DM);   // dk 32..63

  const int ktcnt = q0blk / 64 + 2;                                 // block-uniform
  const float scale = 0.125f;                                       // 1/sqrt(64)
  float m8[8], l8[8];
#pragma unroll
  for (int i = 0; i < 8; ++i) { m8[i] = -1e30f; l8[i] = 0.f; }

  // ---- pass 1: online row max / sum ----
#if HAVE_TDM
  if (wave == 0) tdm_load_tile(lds_off(&kstage[0][0]), kb, DM);
#endif
  for (int kt = 0; kt < ktcnt; ++kt) {
    const int kk0 = kt * 64;
#if HAVE_TDM
    const unsigned short* klds = &kstage[kt & 1][0];
    if (wave == 0) {
      if (kt + 1 < ktcnt) {
        tdm_load_tile(lds_off(&kstage[(kt + 1) & 1][0]), kb + (size_t)(kk0 + 64) * DM, DM);
        __builtin_amdgcn_s_wait_tensorcnt(1);
      } else {
        __builtin_amdgcn_s_wait_tensorcnt(0);
      }
    }
#else
    const unsigned short* klds = &kstage[0][0];
    stage_tile(&kstage[0][0], kb + (size_t)kk0 * DM, DM);
#endif
    __syncthreads();
    v8f S[4];
#pragma unroll
    for (int n = 0; n < 4; ++n) {
      v16bf b0 = load_frag_b16(klds + (n * 16) * 72, 72);
      v16bf b1 = load_frag_b16(klds + (n * 16) * 72 + 32, 72);
      v8f c = {};
      c = WMMA(qlo, b0, c);
      c = WMMA(qhi, b1, c);
      S[n] = c;
    }
#pragma unroll
    for (int rr = 0; rr < 8; ++rr) {
      const int rowg = q0 + rr + half * 8;
      float sv[4], smax = -1e30f;
#pragma unroll
      for (int n = 0; n < 4; ++n) {
        const int colg = kk0 + n * 16 + r;
        float s = S[n][rr] * scale;
        if (colg > rowg) s = -1e30f;
        sv[n] = s;
        smax = fmaxf(smax, s);
      }
#pragma unroll
      for (int d = 1; d < 16; d <<= 1) smax = fmaxf(smax, __shfl_xor(smax, d, 32));
      const float mnew = fmaxf(m8[rr], smax);
      float es = 0.f;
#pragma unroll
      for (int n = 0; n < 4; ++n) es += __expf(sv[n] - mnew);
#pragma unroll
      for (int d = 1; d < 16; d <<= 1) es += __shfl_xor(es, d, 32);
      l8[rr] = l8[rr] * __expf(m8[rr] - mnew) + es;
      m8[rr] = mnew;
    }
    __syncthreads();
  }
  float invl[8];
#pragma unroll
  for (int i = 0; i < 8; ++i) invl[i] = 1.f / l8[i];

  // ---- pass 2: recompute scores, emit pattern, accumulate z = P @ V ----
  v8f zv = {};
  v8f zacc[4] = {zv, zv, zv, zv};
#if HAVE_TDM
  if (wave == 0) {
    tdm_load_tile(lds_off(&kstage[0][0]), kb, DM);
    tdm_load_tile(lds_off(&vstage[0][0]), vb, PSEQ);
  }
#endif
  for (int kt = 0; kt < ktcnt; ++kt) {
    const int kk0 = kt * 64;
#if HAVE_TDM
    const unsigned short* klds = &kstage[kt & 1][0];
    const unsigned short* vlds = &vstage[kt & 1][0];
    if (wave == 0) {
      if (kt + 1 < ktcnt) {
        tdm_load_tile(lds_off(&kstage[(kt + 1) & 1][0]), kb + (size_t)(kk0 + 64) * DM, DM);
        tdm_load_tile(lds_off(&vstage[(kt + 1) & 1][0]), vb + kk0 + 64, PSEQ);
        __builtin_amdgcn_s_wait_tensorcnt(2);
      } else {
        __builtin_amdgcn_s_wait_tensorcnt(0);
      }
    }
#else
    const unsigned short* klds = &kstage[0][0];
    const unsigned short* vlds = &vstage[0][0];
    stage_tile(&kstage[0][0], kb + (size_t)kk0 * DM, DM);
    stage_tile(&vstage[0][0], vb + kk0, PSEQ);
#endif
    __syncthreads();
    v8f S[4];
#pragma unroll
    for (int n = 0; n < 4; ++n) {
      v16bf b0 = load_frag_b16(klds + (n * 16) * 72, 72);
      v16bf b1 = load_frag_b16(klds + (n * 16) * 72 + 32, 72);
      v8f c = {};
      c = WMMA(qlo, b0, c);
      c = WMMA(qhi, b1, c);
      S[n] = c;
    }
#pragma unroll
    for (int n = 0; n < 4; ++n) {
#pragma unroll
      for (int rr = 0; rr < 8; ++rr) {
        const int rowl = rr + half * 8;
        const int rowg = q0 + rowl;
        const int colg = kk0 + n * 16 + r;
        float pv = __expf(S[n][rr] * scale - m8[rr]) * invl[rr];
        if (colg > rowg) pv = 0.f;
        patb[(size_t)rowl * PSEQ + colg] = pv;
        plds[wave][rowl][n * 16 + r] = (unsigned short)bfbits(pv);
      }
    }
    __builtin_amdgcn_wave_barrier();
    v16bf a0 = load_frag_b16(&plds[wave][0][0], 88);
    v16bf a1 = load_frag_b16(&plds[wave][0][32], 88);
#pragma unroll
    for (int n = 0; n < 4; ++n) {
      v16bf b0 = load_frag_b16(vlds + (n * 16) * 72, 72);
      v16bf b1 = load_frag_b16(vlds + (n * 16) * 72 + 32, 72);
      zacc[n] = WMMA(a0, b0, zacc[n]);
      zacc[n] = WMMA(a1, b1, zacc[n]);
    }
    __syncthreads();
  }

  // ---- zero-fill masked columns [ktcnt*64, PSEQ) ----
  const int cstart = ktcnt * 64;
  const float4 f4z = make_float4(0.f, 0.f, 0.f, 0.f);
#pragma unroll 1
  for (int row = 0; row < 16; ++row) {
    float* pr = patb + (size_t)row * PSEQ;
    for (int c = cstart + lane * 4; c < PSEQ; c += 128)
      *(float4*)(pr + c) = f4z;
  }

  // ---- store z (bf16, [b,p,h,dk]) ----
  unsigned short* zb = zout + ((size_t)bb * PSEQ + q0) * DM + h * DH;
#pragma unroll
  for (int n = 0; n < 4; ++n)
#pragma unroll
    for (int rr = 0; rr < 8; ++rr)
      zb[(size_t)(rr + half * 8) * DM + n * 16 + r] = (unsigned short)bfbits(zacc[n][rr]);
}

// -------- output projection: out[m,d] = sum_c z[m,c] * OT[d,c] + Ob[d] --------
__global__ __launch_bounds__(256)
void k_outproj(const unsigned short* __restrict__ z, const unsigned short* __restrict__ OT,
               const float* __restrict__ Ob, float* __restrict__ out) {
  const int d0   = blockIdx.y * 64;
  const int wave = threadIdx.x >> 5, lane = threadIdx.x & 31;
  const int r    = lane & 15,        half = lane >> 4;
  const int m0   = blockIdx.x * 128 + wave * 16;
  v8f zv = {};
  v8f acc[4] = {zv, zv, zv, zv};
  for (int k0 = 0; k0 < DM; k0 += 32) {
    __builtin_prefetch(z + (size_t)m0 * DM + k0 + 32, 0, 3);
    v16bf a = load_frag_b16(z + (size_t)m0 * DM + k0, DM);
#pragma unroll
    for (int n = 0; n < 4; ++n) {
      v16bf b = load_frag_b16(OT + (size_t)(d0 + n * 16) * DM + k0, DM);
      acc[n] = WMMA(a, b, acc[n]);
    }
  }
#pragma unroll
  for (int n = 0; n < 4; ++n) {
    const float bv = Ob[d0 + n * 16 + r];
#pragma unroll
    for (int rr = 0; rr < 8; ++rr)
      out[(size_t)(m0 + rr + half * 8) * DM + d0 + n * 16 + r] = acc[n][rr] + bv;
  }
}

extern "C" void kernel_launch(void* const* d_in, const int* in_sizes, int n_in,
                              void* d_out, int out_size, void* d_ws, size_t ws_size,
                              hipStream_t stream) {
  const float* resid = (const float*)d_in[0];   // normalized_resid_pre [B,P,DM]
  const float* enc   = (const float*)d_in[1];   // encoder_output      [B,P,DM]
  const float* Qs    = (const float*)d_in[2];   // [H,DM,DH]
  const float* Ks    = (const float*)d_in[3];
  const float* Vs    = (const float*)d_in[4];
  const float* O     = (const float*)d_in[5];   // [H,DH,DM]
  const float* Qbs   = (const float*)d_in[6];
  const float* Kbs   = (const float*)d_in[7];
  const float* Vbs   = (const float*)d_in[8];
  const float* Ob    = (const float*)d_in[9];

  const size_t ACT = (size_t)NB * PSEQ * DM;    // 4,194,304 elems
  const size_t WEL = (size_t)NH * DM * DH;      // 1,048,576 elems

  unsigned short* ws    = (unsigned short*)d_ws;
  unsigned short* q     = ws;                   // [b,p,h,dk] bf16
  unsigned short* k     = q + ACT;              // [b,p,h,dk]
  unsigned short* vT    = k + ACT;              // [b,h,dk,p]
  unsigned short* zb    = vT + ACT;             // [b,p,h,dk]
  unsigned short* encbf = zb + ACT;             // [b,p,d]
  unsigned short* resbf = encbf + ACT;          // [b,p,d]
  unsigned short* wtq   = resbf + ACT;          // [h,dk,d]
  unsigned short* wtk   = wtq + WEL;
  unsigned short* wtv   = wtk + WEL;
  unsigned short* ot    = wtv + WEL;            // [d, h*64+dk]

  float* attn_out = (float*)d_out;
  float* pattern  = attn_out + ACT;

  k_cvt_bf16<<<dim3(ACT / 2048), 256, 0, stream>>>(enc, encbf);
  k_cvt_bf16<<<dim3(ACT / 2048), 256, 0, stream>>>(resid, resbf);

  dim3 tb(32, 8);
  k_transpose_cvt<<<dim3(2, 32, NH), tb, 0, stream>>>(Qs, wtq, DM, DH);
  k_transpose_cvt<<<dim3(2, 32, NH), tb, 0, stream>>>(Ks, wtk, DM, DH);
  k_transpose_cvt<<<dim3(2, 32, NH), tb, 0, stream>>>(Vs, wtv, DM, DH);
  k_transpose_cvt<<<dim3(32, 32, 1), tb, 0, stream>>>(O, ot, DM, DM);

  k_proj<<<dim3(32, NH), 256, 0, stream>>>(encbf, wtq, Qbs, q,  0);
  k_proj<<<dim3(32, NH), 256, 0, stream>>>(encbf, wtk, Kbs, k,  0);
  k_proj<<<dim3(32, NH), 256, 0, stream>>>(resbf, wtv, Vbs, vT, 1);

  k_attn<<<dim3(PSEQ / 128, NB * NH), 256, 0, stream>>>(q, k, vT, zb, pattern);

  k_outproj<<<dim3(32, DM / 64), 256, 0, stream>>>(zb, ot, Ob, attn_out);
}
